// BinaryLinear_76493367542676
// MI455X (gfx1250) — 8-bit matrix operand order repaired and checked by reading, compiled, not run on hardware
//
#include <hip/hip_runtime.h>
#include <hip/hip_bf16.h>

typedef __attribute__((ext_vector_type(8))) int v8i;

#define NROWS 8192
#define KDIM  4096
#define ODIM  4096

__global__ void abs_sum_kernel(const float* __restrict__ p, long long n,
                               float* __restrict__ out) {
    long long i = (long long)blockIdx.x * blockDim.x + threadIdx.x;
    const long long stride = (long long)gridDim.x * blockDim.x;
    float s = 0.f;
    for (; i < n; i += stride) s += fabsf(p[i]);
    #pragma unroll
    for (int off = 16; off > 0; off >>= 1)
        s += __shfl_xor(s, off, 32);
    __shared__ float lds[8];
    const int lane = threadIdx.x & 31;
    const int wv   = threadIdx.x >> 5;
    if (lane == 0) lds[wv] = s;
    __syncthreads();
    if (threadIdx.x < 8) {
        float t = lds[threadIdx.x];
        #pragma unroll
        for (int off = 4; off > 0; off >>= 1)
            t += __shfl_xor(t, off, 32);
        if (threadIdx.x == 0) atomicAdd(out, t);
    }
}

__global__ void pack_sign_kernel(const float* __restrict__ src,
                                 int* __restrict__ dst, long long n4) {
    long long i = (long long)blockIdx.x * blockDim.x + threadIdx.x;
    const long long stride = (long long)gridDim.x * blockDim.x;
    for (; i < n4; i += stride) {
        const float4 v = ((const float4*)src)[i];
        const int s0 = (v.x > 0.f) - (v.x < 0.f);
        const int s1 = (v.y > 0.f) - (v.y < 0.f);
        const int s2 = (v.z > 0.f) - (v.z < 0.f);
        const int s3 = (v.w > 0.f) - (v.w < 0.f);
        dst[i] = (s0 & 0xFF) | ((s1 & 0xFF) << 8) |
                 ((s2 & 0xFF) << 16) | ((s3 & 0xFF) << 24);
    }
}

__device__ __forceinline__ v8i load_frag_a(const signed char* p) {
    v8i f;
    const int2 t0 = *(const int2*)(p +  0);
    const int2 t1 = *(const int2*)(p +  8);
    const int2 t2 = *(const int2*)(p + 32);
    const int2 t3 = *(const int2*)(p + 40);
    f[0] = t0.x; f[1] = t0.y; f[2] = t1.x; f[3] = t1.y;
    f[4] = t2.x; f[5] = t2.y; f[6] = t3.x; f[7] = t3.y;
    return f;
}

__device__ __forceinline__ v8i load_frag_b(const signed char* p) {
    v8i f;
    const int4 t0 = *(const int4*)(p +  0);
    const int4 t1 = *(const int4*)(p + 32);
    f[0] = t0.x; f[1] = t0.y; f[2] = t0.z; f[3] = t0.w;
    f[4] = t1.x; f[5] = t1.y; f[6] = t1.z; f[7] = t1.w;
    return f;
}

__global__ __launch_bounds__(256) void binlin_wmma_kernel(
    const signed char* __restrict__ sA,
    const signed char* __restrict__ sB,
    const float* __restrict__ sums,
    const float* __restrict__ bias,
    float* __restrict__ out) {
    const int lane  = threadIdx.x & 31;
    const int wv    = threadIdx.x >> 5;
    const int waveM = wv >> 1;
    const int waveN = wv & 1;
    const int m0 = blockIdx.x * 128 + waveM * 32;
    const int n0 = blockIdx.y * 128 + waveN * 64;
    const int l15 = lane & 15;
    const int lhi = lane >> 4;

    const signed char* pa0 = sA + (size_t)(m0 + l15) * KDIM + lhi * 16;
    const signed char* pa1 = pa0 + (size_t)16 * KDIM;
    const signed char* pb0 = sB + (size_t)(n0 + l15) * KDIM + lhi * 16;
    const signed char* pb1 = pb0 + (size_t)16 * KDIM;
    const signed char* pb2 = pb0 + (size_t)32 * KDIM;
    const signed char* pb3 = pb0 + (size_t)48 * KDIM;

    v8i acc[2][4];
    #pragma unroll
    for (int mt = 0; mt < 2; ++mt)
        #pragma unroll
        for (int nt = 0; nt < 4; ++nt)
            acc[mt][nt] = (v8i){};

    for (int k = 0; k < KDIM; k += 64) {
        const v8i a0 = load_frag_a(pa0 + k);
        const v8i a1 = load_frag_a(pa1 + k);
        const v8i b0 = load_frag_b(pb0 + k);
        const v8i b1 = load_frag_b(pb1 + k);
        const v8i b2 = load_frag_b(pb2 + k);
        const v8i b3 = load_frag_b(pb3 + k);
        acc[0][0] = __builtin_amdgcn_wmma_i32_16x16x64_iu8(true, a0, true, b0, acc[0][0], false, false);
        acc[0][1] = __builtin_amdgcn_wmma_i32_16x16x64_iu8(true, a0, true, b1, acc[0][1], false, false);
        acc[0][2] = __builtin_amdgcn_wmma_i32_16x16x64_iu8(true, a0, true, b2, acc[0][2], false, false);
        acc[0][3] = __builtin_amdgcn_wmma_i32_16x16x64_iu8(true, a0, true, b3, acc[0][3], false, false);
        acc[1][0] = __builtin_amdgcn_wmma_i32_16x16x64_iu8(true, a1, true, b0, acc[1][0], false, false);
        acc[1][1] = __builtin_amdgcn_wmma_i32_16x16x64_iu8(true, a1, true, b1, acc[1][1], false, false);
        acc[1][2] = __builtin_amdgcn_wmma_i32_16x16x64_iu8(true, a1, true, b2, acc[1][2], false, false);
        acc[1][3] = __builtin_amdgcn_wmma_i32_16x16x64_iu8(true, a1, true, b3, acc[1][3], false, false);
    }

    const float s_x = sums[0] * (1.0f / ((float)NROWS * (float)KDIM));
    const float s_w = sums[1] * (1.0f / ((float)ODIM  * (float)KDIM));
    const float s_b = sums[2] * (1.0f / (float)ODIM);
    const float scale = s_x * s_w;

    const int mrow = m0 + lhi * 8;
    #pragma unroll
    for (int nt = 0; nt < 4; ++nt) {
        const int col = n0 + nt * 16 + l15;
        const float bv = bias[col];
        const float bt = s_b * (float)((bv > 0.f) - (bv < 0.f));
        #pragma unroll
        for (int r = 0; r < 8; ++r) {
            out[(size_t)(mrow + r)      * ODIM + col] = scale * (float)acc[0][nt][r] + bt;
            out[(size_t)(mrow + 16 + r) * ODIM + col] = scale * (float)acc[1][nt][r] + bt;
        }
    }
}

extern "C" void kernel_launch(void* const* d_in, const int* in_sizes, int n_in,
                              void* d_out, int out_size, void* d_ws, size_t ws_size,
                              hipStream_t stream) {
    const float* x = (const float*)d_in[0];
    const float* w = (const float*)d_in[1];
    const float* b = (const float*)d_in[2];
    float* out = (float*)d_out;

    char* ws = (char*)d_ws;
    float* sums = (float*)ws;
    signed char* sA = (signed char*)(ws + 64);
    signed char* sB = sA + (size_t)NROWS * KDIM;

    hipMemsetAsync(ws, 0, 64, stream);

    abs_sum_kernel<<<512, 256, 0, stream>>>(x, (long long)NROWS * KDIM, sums + 0);
    abs_sum_kernel<<<512, 256, 0, stream>>>(w, (long long)ODIM  * KDIM, sums + 1);
    abs_sum_kernel<<<16,  256, 0, stream>>>(b, (long long)ODIM,         sums + 2);

    pack_sign_kernel<<<1024, 256, 0, stream>>>(x, (int*)sA, (long long)NROWS * KDIM / 4);
    pack_sign_kernel<<<1024, 256, 0, stream>>>(w, (int*)sB, (long long)ODIM  * KDIM / 4);

    dim3 grid(NROWS / 128, ODIM / 128);
    binlin_wmma_kernel<<<grid, 256, 0, stream>>>(sA, sB, sums, b, out);
}
